// QCDTransAttention_52140902974005
// MI455X (gfx1250) — matrix operand order repaired and checked by reading, not compiled, not run on hardware
//
#include <hip/hip_runtime.h>
#include <hip/hip_bf16.h>
#include <stdint.h>

#define EPSF 1.1920929e-07f

typedef __attribute__((ext_vector_type(16))) _Float16 v16h;
typedef __attribute__((ext_vector_type(8)))  _Float16 v8h;
typedef __attribute__((ext_vector_type(8)))  float    v8f;
union AV16 { v16h v; v8h h[2]; };

// ---- geometry ----
#define Bq 64
#define Nq 197
#define Cq 768
#define Hq 12
#define HD 64
#define BH (Bq*Hq)        // 768 (batch*heads)
#define NPAD 208          // 13*16  padded token rows
#define KPAD 224          // 14*16  padded K/col dim for attn & vT
#define M_ROWS (Bq*Nq)    // 12608 (multiple of 16)
#define NKV (3*Cq)        // 2304

__device__ __forceinline__ v8f wmma_f16(v16h a, v16h b, v8f c) {
  // v_wmma_f32_16x16x32_f16: D = A(16x32 f16) * B(32x16 f16) + C(16x16 f32)
  return __builtin_amdgcn_wmma_f32_16x16x32_f16(false, a, false, b, (short)0, c,
                                                false, false);
}

__device__ __forceinline__ void load_a(AV16& a, const _Float16* arow, int k0, int kh) {
  a.h[0] = *(const v8h*)(arow + k0 + kh * 16);
  a.h[1] = *(const v8h*)(arow + k0 + kh * 16 + 8);
}
__device__ __forceinline__ void load_b(AV16& b, const _Float16* brow, int k0) {
  b.h[0] = *(const v8h*)(brow + k0);
  b.h[1] = *(const v8h*)(brow + k0 + 8);
}

// ---------------- elementwise / utility kernels ----------------

__global__ void k_fill_u32(uint32_t* p, int n) {
  int i = blockIdx.x * 256 + threadIdx.x;
  if (i < n) p[i] = 0u;
}

__global__ void k_cvt_f16(const float* __restrict__ s, _Float16* __restrict__ d, int n) {
  int i = blockIdx.x * 256 + threadIdx.x;
  if (i < n) d[i] = (_Float16)s[i];
}

// per-row symmetric int8 fake-quant of weights, emit f16 (one block per row)
__global__ void k_fq_weight(const float* __restrict__ w, _Float16* __restrict__ o, int cols) {
  int row = blockIdx.x;
  int t = threadIdx.x;
  const float* wr = w + (size_t)row * cols;
  __shared__ float smn[256], smx[256];
  float mn = 3.0e38f, mx = -3.0e38f;
  for (int c = t; c < cols; c += 256) {
    float v = wr[c];
    mn = fminf(mn, v);
    mx = fmaxf(mx, v);
  }
  smn[t] = mn; smx[t] = mx;
  __syncthreads();
  for (int st = 128; st > 0; st >>= 1) {
    if (t < st) { smn[t] = fminf(smn[t], smn[t + st]); smx[t] = fmaxf(smx[t], smx[t + st]); }
    __syncthreads();
  }
  float scale = fmaxf(fmaxf(-smn[0], smx[0]) * (1.0f / 127.5f), EPSF);
  _Float16* orow = o + (size_t)row * cols;
  for (int c = t; c < cols; c += 256) {
    float q = fminf(fmaxf(rintf(wr[c] / scale), -128.0f), 127.0f);
    orow[c] = (_Float16)(q * scale);
  }
}

// ---------------- WMMA GEMM kernels ----------------
// One wave computes a 16(M) x 64(N) strip: 4 accumulators, A fragment reused 4x.

// out = x16 @ wq16^T + b ; scattered into q16/k16 [bh][t(208)][64] and vT16 [bh][d][t(224)]
__global__ void k_gemm_qkv(const _Float16* __restrict__ x16, const _Float16* __restrict__ w16,
                           const float* __restrict__ bias,
                           _Float16* __restrict__ q16, _Float16* __restrict__ k16,
                           _Float16* __restrict__ vt16) {
  const int lane = threadIdx.x;
  const int nl = lane & 15, kh = lane >> 4;
  const int n0 = blockIdx.x * 64;
  const int m0 = blockIdx.y * 16;
  const _Float16* arow = x16 + (size_t)(m0 + nl) * Cq;
  const _Float16* brow0 = w16 + (size_t)(n0 + nl) * Cq + kh * 16;
  const _Float16* brow1 = brow0 + (size_t)16 * Cq;
  const _Float16* brow2 = brow0 + (size_t)32 * Cq;
  const _Float16* brow3 = brow0 + (size_t)48 * Cq;
  v8f acc0 = {}, acc1 = {}, acc2 = {}, acc3 = {};
#pragma unroll 2
  for (int k0 = 0; k0 < Cq; k0 += 32) {
    AV16 a, b0, b1, b2, b3;
    load_a(a, arow, k0, kh);
    load_b(b0, brow0, k0);
    load_b(b1, brow1, k0);
    load_b(b2, brow2, k0);
    load_b(b3, brow3, k0);
    acc0 = wmma_f16(a.v, b0.v, acc0);
    acc1 = wmma_f16(a.v, b1.v, acc1);
    acc2 = wmma_f16(a.v, b2.v, acc2);
    acc3 = wmma_f16(a.v, b3.v, acc3);
  }
  // 64-col group is aligned: constant (which, h) per block
  const int which = n0 / Cq;          // 0=q 1=k 2=v
  const int h = (n0 - which * Cq) >> 6;
  v8f accs[4] = {acc0, acc1, acc2, acc3};
#pragma unroll
  for (int i = 0; i < 4; ++i) {
    const int d = i * 16 + nl;
    const float bv = bias[n0 + i * 16 + nl];
#pragma unroll
    for (int j = 0; j < 8; ++j) {
      int m = m0 + j + 8 * kh;
      int bb = m / Nq;
      int tkn = m - bb * Nq;
      int bh = bb * Hq + h;
      float val = accs[i][j] + bv;
      if (which == 0)      q16[((size_t)bh * NPAD + tkn) * HD + d]  = (_Float16)val;
      else if (which == 1) k16[((size_t)bh * NPAD + tkn) * HD + d]  = (_Float16)val;
      else                 vt16[((size_t)bh * HD + d) * KPAD + tkn] = (_Float16)val;
    }
  }
}

// attnf16[bh][m][n] = 0.125 * q[bh][m][:] . k[bh][n][:]   (padded, pads give exact 0)
__global__ void k_gemm_qk(const _Float16* __restrict__ q16, const _Float16* __restrict__ k16,
                          _Float16* __restrict__ attnf) {
  const int lane = threadIdx.x;
  const int nl = lane & 15, kh = lane >> 4;
  const int n0 = blockIdx.x * 16;
  const int m0 = blockIdx.y * 16;
  const int bh = blockIdx.z;
  const _Float16* arow = q16 + ((size_t)bh * NPAD + m0 + nl) * HD;
  const _Float16* brow = k16 + ((size_t)bh * NPAD + n0 + nl) * HD + kh * 16;
  v8f acc = {};
#pragma unroll
  for (int k0 = 0; k0 < HD; k0 += 32) {
    AV16 a, b;
    load_a(a, arow, k0, kh);
    load_b(b, brow, k0);
    acc = wmma_f16(a.v, b.v, acc);
  }
  _Float16* op = attnf + ((size_t)bh * NPAD + m0) * KPAD + n0 + nl;
#pragma unroll
  for (int j = 0; j < 8; ++j) op[(size_t)(j + 8 * kh) * KPAD] = (_Float16)(acc[j] * 0.125f);
}

// ctx[b*197+t][h*64 + d] = attn16[bh][t][:224] . vT[bh][d][:224]  (all 64 d per wave)
__global__ void k_gemm_av(const _Float16* __restrict__ a16, const _Float16* __restrict__ vt16,
                          float* __restrict__ ctx) {
  const int lane = threadIdx.x;
  const int nl = lane & 15, kh = lane >> 4;
  const int m0 = blockIdx.x * 16;   // token tile
  const int bh = blockIdx.y;
  const _Float16* arow = a16 + ((size_t)bh * NPAD + m0 + nl) * KPAD;
  const _Float16* brow0 = vt16 + ((size_t)bh * HD + nl) * KPAD + kh * 16;
  const _Float16* brow1 = brow0 + (size_t)16 * KPAD;
  const _Float16* brow2 = brow0 + (size_t)32 * KPAD;
  const _Float16* brow3 = brow0 + (size_t)48 * KPAD;
  v8f acc0 = {}, acc1 = {}, acc2 = {}, acc3 = {};
#pragma unroll
  for (int k0 = 0; k0 < KPAD; k0 += 32) {
    AV16 a, b0, b1, b2, b3;
    load_a(a, arow, k0, kh);
    load_b(b0, brow0, k0);
    load_b(b1, brow1, k0);
    load_b(b2, brow2, k0);
    load_b(b3, brow3, k0);
    acc0 = wmma_f16(a.v, b0.v, acc0);
    acc1 = wmma_f16(a.v, b1.v, acc1);
    acc2 = wmma_f16(a.v, b2.v, acc2);
    acc3 = wmma_f16(a.v, b3.v, acc3);
  }
  const int bb = bh / Hq, h = bh - bb * Hq;
  v8f accs[4] = {acc0, acc1, acc2, acc3};
#pragma unroll
  for (int j = 0; j < 8; ++j) {
    int tkn = m0 + j + 8 * kh;
    if (tkn < Nq) {
      float* op = ctx + ((size_t)(bb * Nq + tkn)) * Cq + h * HD + nl;
#pragma unroll
      for (int i = 0; i < 4; ++i) op[i * 16] = accs[i][j];
    }
  }
}

// projf = ctx16 @ wp16^T + b_proj  (dense f32 out)
__global__ void k_gemm_proj(const _Float16* __restrict__ a, const _Float16* __restrict__ w,
                            const float* __restrict__ bias, float* __restrict__ out) {
  const int lane = threadIdx.x;
  const int nl = lane & 15, kh = lane >> 4;
  const int n0 = blockIdx.x * 64;
  const int m0 = blockIdx.y * 16;
  const _Float16* arow = a + (size_t)(m0 + nl) * Cq;
  const _Float16* brow0 = w + (size_t)(n0 + nl) * Cq + kh * 16;
  const _Float16* brow1 = brow0 + (size_t)16 * Cq;
  const _Float16* brow2 = brow0 + (size_t)32 * Cq;
  const _Float16* brow3 = brow0 + (size_t)48 * Cq;
  v8f acc0 = {}, acc1 = {}, acc2 = {}, acc3 = {};
#pragma unroll 2
  for (int k0 = 0; k0 < Cq; k0 += 32) {
    AV16 av, b0, b1, b2, b3;
    load_a(av, arow, k0, kh);
    load_b(b0, brow0, k0);
    load_b(b1, brow1, k0);
    load_b(b2, brow2, k0);
    load_b(b3, brow3, k0);
    acc0 = wmma_f16(av.v, b0.v, acc0);
    acc1 = wmma_f16(av.v, b1.v, acc1);
    acc2 = wmma_f16(av.v, b2.v, acc2);
    acc3 = wmma_f16(av.v, b3.v, acc3);
  }
  v8f accs[4] = {acc0, acc1, acc2, acc3};
#pragma unroll
  for (int i = 0; i < 4; ++i) {
    const float bb = bias[n0 + i * 16 + nl];
    float* op = out + (size_t)m0 * Cq + n0 + i * 16 + nl;
#pragma unroll
    for (int j = 0; j < 8; ++j) op[(size_t)(j + 8 * kh) * Cq] = accs[i][j] + bb;
  }
}

// ---------------- reductions + quantization ----------------

__global__ void k_minmax_attn(const _Float16* __restrict__ attnf, float* __restrict__ part) {
  const int total = BH * Nq * Nq;
  const int t = threadIdx.x;
  int stride = gridDim.x * 256;
  float mn = 3.0e38f, mx = -3.0e38f;
  for (int i = blockIdx.x * 256 + t; i < total; i += stride) {
    int bh = i / (Nq * Nq);
    int rem = i - bh * (Nq * Nq);
    int r = rem / Nq;
    int c = rem - r * Nq;
    float v = (float)attnf[((size_t)bh * NPAD + r) * KPAD + c];
    mn = fminf(mn, v);
    mx = fmaxf(mx, v);
  }
  __shared__ float smn[256], smx[256];
  smn[t] = mn; smx[t] = mx;
  __syncthreads();
  for (int st = 128; st > 0; st >>= 1) {
    if (t < st) { smn[t] = fminf(smn[t], smn[t + st]); smx[t] = fmaxf(smx[t], smx[t + st]); }
    __syncthreads();
  }
  if (t == 0) { part[2 * blockIdx.x] = smn[0]; part[2 * blockIdx.x + 1] = smx[0]; }
}

__global__ void k_minmax_dense(const float* __restrict__ src, int n, float* __restrict__ part) {
  const int t = threadIdx.x;
  int stride = gridDim.x * 256;
  float mn = 3.0e38f, mx = -3.0e38f;
  for (int i = blockIdx.x * 256 + t; i < n; i += stride) {
    float v = src[i];
    mn = fminf(mn, v);
    mx = fmaxf(mx, v);
  }
  __shared__ float smn[256], smx[256];
  smn[t] = mn; smx[t] = mx;
  __syncthreads();
  for (int st = 128; st > 0; st >>= 1) {
    if (t < st) { smn[t] = fminf(smn[t], smn[t + st]); smx[t] = fmaxf(smx[t], smx[t + st]); }
    __syncthreads();
  }
  if (t == 0) { part[2 * blockIdx.x] = smn[0]; part[2 * blockIdx.x + 1] = smx[0]; }
}

// reduce partials -> uint4 asymmetric (scale, zero_point)
__global__ void k_minmax_final(const float* __restrict__ part, int nblk, float* __restrict__ sc) {
  const int t = threadIdx.x;
  float mn = 3.0e38f, mx = -3.0e38f;
  for (int i = t; i < nblk; i += 256) {
    mn = fminf(mn, part[2 * i]);
    mx = fmaxf(mx, part[2 * i + 1]);
  }
  __shared__ float smn[256], smx[256];
  smn[t] = mn; smx[t] = mx;
  __syncthreads();
  for (int st = 128; st > 0; st >>= 1) {
    if (t < st) { smn[t] = fminf(smn[t], smn[t + st]); smx[t] = fmaxf(smx[t], smx[t + st]); }
    __syncthreads();
  }
  if (t == 0) {
    float scale = fmaxf((smx[0] - smn[0]) * (1.0f / 15.0f), EPSF);
    sc[0] = scale;
    sc[1] = -rintf(smn[0] / scale);   // zp = qmin(0) - round(mn/scale)
  }
}

__global__ void k_quant_u4_f16(const float* __restrict__ s, _Float16* __restrict__ d,
                               const float* __restrict__ sc, int n) {
  int i = blockIdx.x * 256 + threadIdx.x;
  if (i >= n) return;
  float scale = sc[0], zp = sc[1];
  float q = fminf(fmaxf(rintf(s[i] / scale) + zp, 0.0f), 15.0f);
  d[i] = (_Float16)((q - zp) * scale);
}

__global__ void k_quant_u4_f32(const float* __restrict__ s, float* __restrict__ d,
                               const float* __restrict__ sc, int n) {
  int i = blockIdx.x * 256 + threadIdx.x;
  if (i >= n) return;
  float scale = sc[0], zp = sc[1];
  float q = fminf(fmaxf(rintf(s[i] / scale) + zp, 0.0f), 15.0f);
  d[i] = (q - zp) * scale;
}

// ---------------- FQ-ViT int log2-softmax (one block per padded row) ----------------
__global__ void k_softmax(const _Float16* __restrict__ attnf, _Float16* __restrict__ a16,
                          const float* __restrict__ sc) {
  const int bh = blockIdx.x;
  const int r = blockIdx.y;
  const int t = threadIdx.x;       // 256 threads, cols 0..223 written
  const size_t row = ((size_t)bh * NPAD + r) * KPAD;
  if (r >= Nq) {                   // pad row -> zeros
    if (t < KPAD) a16[row + t] = (_Float16)0.0f;
    return;
  }
  const float s = sc[0], zp = sc[1];
  const bool valid = t < Nq;
  float xint = -3.0e38f;
  if (valid) {
    float a = (float)attnf[row + t];
    float q = fminf(fmaxf(rintf(a / s) + zp, 0.0f), 15.0f);   // uint4 fake-quant
    xint = q - zp;                 // == dequant / scaling_factor, exactly
  }
  __shared__ float sh[256];
  sh[t] = xint;
  __syncthreads();
  for (int st = 128; st > 0; st >>= 1) {
    if (t < st) sh[t] = fmaxf(sh[t], sh[t + st]);
    __syncthreads();
  }
  const float mx = sh[0];
  __syncthreads();
  const float x0i = floorf(-0.6931f / s);
  const float bi  = floorf((0.96963238f / 0.35815147f) / s);
  const float ci  = floorf((1.0f / 0.35815147f) / (s * s));
  float expi = 0.0f;
  if (valid) {
    float xi = fmaxf(xint - mx, 30.0f * x0i);
    float qf = floorf(xi / x0i);
    float rr = xi - x0i * qf;
    float z  = rr * (rr + bi) + ci;
    expi = fmaxf(floorf(z * exp2f(30.0f - qf)), 0.0f);
  }
  sh[t] = expi;
  __syncthreads();
  for (int st = 128; st > 0; st >>= 1) {
    if (t < st) sh[t] = sh[t] + sh[t + st];
    __syncthreads();
  }
  const float sum = sh[0];
  float out = 0.0f;
  if (valid && expi > 0.0f) {
    float ratio = rintf(sum / fmaxf(expi, 1.0f));
    float big = floorf(log2f(ratio));
    float rounds = (ratio - exp2f(big) >= exp2f(big - 1.0f)) ? big + 1.0f : big;
    out = (rounds >= 256.0f) ? 0.0f : exp2f(-fminf(fmaxf(rounds, 0.0f), 255.0f));
  }
  if (t < KPAD) a16[row + t] = (_Float16)out;   // pad cols (>=197) -> 0
}

// ---------------- host ----------------

extern "C" void kernel_launch(void* const* d_in, const int* in_sizes, int n_in,
                              void* d_out, int out_size, void* d_ws, size_t ws_size,
                              hipStream_t stream) {
  (void)in_sizes; (void)n_in; (void)out_size; (void)ws_size;
  const float* x2     = (const float*)d_in[0];
  const float* w_qkv  = (const float*)d_in[1];
  const float* b_qkv  = (const float*)d_in[2];
  const float* w_proj = (const float*)d_in[3];
  const float* b_proj = (const float*)d_in[4];

  char* ws = (char*)d_ws;
  size_t off = 0;
  auto alloc = [&](size_t bytes) -> char* {
    char* p = ws + off;
    off += (bytes + 255) & ~(size_t)255;
    return p;
  };
  const int NC = M_ROWS * Cq;                         // 9,682,944
  _Float16* x16  = (_Float16*)alloc((size_t)NC * 2);            // 19.4 MB
  _Float16* wq16 = (_Float16*)alloc((size_t)NKV * Cq * 2);      //  3.5 MB
  _Float16* wp16 = (_Float16*)alloc((size_t)Cq * Cq * 2);       //  1.2 MB
  _Float16* q16  = (_Float16*)alloc((size_t)BH * NPAD * HD * 2);// 20.4 MB
  _Float16* k16  = (_Float16*)alloc((size_t)BH * NPAD * HD * 2);// 20.4 MB
  _Float16* vt16 = (_Float16*)alloc((size_t)BH * HD * KPAD * 2);// 22.0 MB
  _Float16* attnf= (_Float16*)alloc((size_t)BH * NPAD * KPAD*2);// 71.6 MB  (f16 logits)
  _Float16* a16  = (_Float16*)alloc((size_t)BH * NPAD * KPAD*2);// 71.6 MB  (softmax probs)
  float*    part = (float*)alloc(1024 * 2 * 4);
  float*    scal = (float*)alloc(256);
  // Aliases (lifetimes disjoint, stream-ordered):
  float*    ctxf  = (float*)attnf;   // attnf dead after softmax; 38.7 MB <= 71.6 MB
  float*    projf = (float*)a16;     // a16 dead after gemm_av;   38.7 MB <= 71.6 MB
  _Float16* ctx16 = x16;             // x16 dead after qkv GEMM

  // 1) weight fake-quant -> f16
  k_fq_weight<<<NKV, 256, 0, stream>>>(w_qkv, wq16, Cq);
  k_fq_weight<<<Cq, 256, 0, stream>>>(w_proj, wp16, Cq);
  // 2) activations -> f16 ; zero q/k/vT pads (contiguous region)
  k_cvt_f16<<<(NC + 255) / 256, 256, 0, stream>>>(x2, x16, NC);
  {
    int nz = (int)(((size_t)BH * NPAD * HD * 2 * 2 + (size_t)BH * HD * KPAD * 2) / 4);
    k_fill_u32<<<(nz + 255) / 256, 256, 0, stream>>>((uint32_t*)q16, nz);
  }
  // 3) qkv GEMM (WMMA, 16x64/wave) -> q16/k16/vT16 with bias
  k_gemm_qkv<<<dim3(NKV / 64, M_ROWS / 16), 32, 0, stream>>>(x16, wq16, b_qkv, q16, k16, vt16);
  // 4) q.k^T (WMMA) -> f16 logits
  k_gemm_qk<<<dim3(NPAD / 16, NPAD / 16, BH), 32, 0, stream>>>(q16, k16, attnf);
  // 5) global minmax of logits -> uint4 scale/zp
  k_minmax_attn<<<1024, 256, 0, stream>>>(attnf, part);
  k_minmax_final<<<1, 256, 0, stream>>>(part, 1024, scal);
  // 6) int log2-softmax -> a16 (f16 powers of two; pads zeroed)
  k_softmax<<<dim3(BH, NPAD), 256, 0, stream>>>(attnf, a16, scal);
  // 7) attn.v (WMMA, 16x64/wave) -> ctxf
  k_gemm_av<<<dim3(NPAD / 16, BH), 32, 0, stream>>>(a16, vt16, ctxf);
  // 8) qact2: global minmax + uint4 quant -> ctx16 (f16)
  k_minmax_dense<<<1024, 256, 0, stream>>>(ctxf, NC, part);
  k_minmax_final<<<1, 256, 0, stream>>>(part, 1024, scal + 2);
  k_quant_u4_f16<<<(NC + 255) / 256, 256, 0, stream>>>(ctxf, ctx16, scal + 2, NC);
  // 9) proj GEMM (WMMA, 16x64/wave) -> projf
  k_gemm_proj<<<dim3(Cq / 64, M_ROWS / 16), 32, 0, stream>>>(ctx16, wp16, b_proj, projf);
  // 10) qact3: global minmax + uint4 quant -> d_out (f32)
  k_minmax_dense<<<1024, 256, 0, stream>>>(projf, NC, part);
  k_minmax_final<<<1, 256, 0, stream>>>(part, 1024, scal + 4);
  k_quant_u4_f32<<<(NC + 255) / 256, 256, 0, stream>>>(projf, (float*)d_out, scal + 4, NC);
}
